// MultiHead_61753039782049
// MI455X (gfx1250) — matrix operand order repaired and checked by reading, not compiled, not run on hardware
//
#include <hip/hip_runtime.h>

#define B_ 4
#define S_ 2048
#define D_ 768
#define H_ 12
#define K_ 64

typedef __attribute__((ext_vector_type(16))) _Float16 v16h;
typedef __attribute__((ext_vector_type(8)))  _Float16 v8h;
typedef __attribute__((ext_vector_type(4)))  _Float16 v4h;
typedef __attribute__((ext_vector_type(8)))  float    v8f;
typedef __attribute__((ext_vector_type(4)))  float    v4f;

static __device__ __forceinline__ v8f wmma_f16(v16h a, v16h b, v8f c) {
  // D = A(16x32 f16) * B(32x16 f16) + C(16x16 f32)
  return __builtin_amdgcn_wmma_f32_16x16x32_f16(
      /*neg_a=*/false, a, /*neg_b=*/false, b,
      /*c_mod=*/(short)0, c, /*reuse_a=*/false, /*reuse_b=*/false);
}

// ---- Fragment loaders (contiguous k order k = 16*(lane/16) + s, wave32) ----

// A-matrix 16x32 f16 from row-major f16 memory (two contiguous 16B chunks/lane).
static __device__ __forceinline__ v16h load_a_f16(const _Float16* base,
                                                  int rowStride, int lane) {
  int row = lane & 15;
  int kh  = (lane >> 4) * 16;
  const _Float16* r = base + row * rowStride;
  v8h lo = *(const v8h*)(r + kh);        // frag elements 0..7
  v8h hi = *(const v8h*)(r + kh + 8);    // frag elements 8..15
  return __builtin_shufflevector(lo, hi, 0,1,2,3,4,5,6,7,8,9,10,11,12,13,14,15);
}

// B-matrix 32x16 where memory is K-major per column: B(k,n) = base[n*stride + k].
// The 16 k-values per lane are contiguous -> single 32B vector load.
static __device__ __forceinline__ v16h load_bT_f16(const _Float16* base,
                                                   int rowStride, int lane) {
  int n  = lane & 15;
  int k0 = (lane >> 4) * 16;
  return *(const v16h*)(base + (size_t)n * rowStride + k0);
}

// Async-stage one contiguous 4KB chunk (32x64 f16) into LDS: 8 x b128 per lane.
static __device__ __forceinline__ void async_stage_4k(const _Float16* gsrc,
                                                      unsigned la) {
#pragma unroll
  for (int i = 0; i < 8; ++i) {
    asm volatile("global_load_async_to_lds_b128 %0, %1, off offset:%2"
                 :: "v"(la), "v"(gsrc), "i"(i * 512) : "memory");
  }
}

// ---------------- Prep 1: x fp32 -> f16 -------------------------------------
__global__ __launch_bounds__(256)
void cvt_x_kernel(const float* __restrict__ x, _Float16* __restrict__ xh) {
  size_t i = ((size_t)blockIdx.x * 256 + threadIdx.x) * 4;
  v4f v = *(const v4f*)(x + i);
  v4h h;
  h[0] = (_Float16)v[0]; h[1] = (_Float16)v[1];
  h[2] = (_Float16)v[2]; h[3] = (_Float16)v[3];
  *(v4h*)(xh + i) = h;
}

// ---------------- Prep 2: W fp32 [h][d][n] -> f16 transposed [which][h][n][d]
__global__ __launch_bounds__(256)
void cvt_w_kernel(const float* __restrict__ Wq, const float* __restrict__ Wk,
                  const float* __restrict__ Wv, _Float16* __restrict__ Wt) {
  int id = blockIdx.x * 256 + threadIdx.x;          // 3*H*K*D total
  int d     = id % D_;
  int n     = (id / D_) % K_;
  int h     = (id / (D_ * K_)) % H_;
  int which = id / (D_ * K_ * H_);
  const float* W = (which == 0) ? Wq : (which == 1) ? Wk : Wv;
  Wt[(size_t)id] = (_Float16)W[(size_t)h * D_ * K_ + (size_t)d * K_ + n];
}

// ---------------- Kernel 1: QKV projection ----------------------------------
// grid = (B*S/64, H, 3), block = 128 (4 waves); each wave owns a 16x64 tile.
// Software-pipelined: iteration i+1's fragments load while i's WMMAs execute.
__global__ __launch_bounds__(128)
void proj_kernel(const _Float16* __restrict__ xh, const _Float16* __restrict__ Wt,
                 _Float16* __restrict__ Qf, _Float16* __restrict__ Kf,
                 _Float16* __restrict__ Vt) {
  int lane = threadIdx.x & 31;
  int wave = threadIdx.x >> 5;
  int rt = blockIdx.x * 4 + wave;
  int h  = blockIdx.y;
  int which = blockIdx.z;

  int gRow = rt * 16;
  int b  = gRow / S_;
  int s0 = gRow % S_;
  const _Float16* xp = xh + ((size_t)b * S_ + s0) * D_;
  const _Float16* Wp = Wt + (size_t)(which * H_ + h) * K_ * D_;   // [n][d], d fastest

  const int NDC = D_ / 32;   // 24 K-steps
  v8f acc0 = {}, acc1 = {}, acc2 = {}, acc3 = {};

  // Prologue: fragments for dc = 0
  v16h a  = load_a_f16(xp, D_, lane);
  v16h b0 = load_bT_f16(Wp, D_, lane);
  v16h b1 = load_bT_f16(Wp + (size_t)16 * D_, D_, lane);
  v16h b2 = load_bT_f16(Wp + (size_t)32 * D_, D_, lane);
  v16h b3 = load_bT_f16(Wp + (size_t)48 * D_, D_, lane);

  for (int i = 0; i < NDC - 1; ++i) {
    int dcn = (i + 1) * 32;
    // Preload next step's fragments into fresh registers.
    v16h an  = load_a_f16(xp + dcn, D_, lane);
    v16h bn0 = load_bT_f16(Wp + dcn, D_, lane);
    v16h bn1 = load_bT_f16(Wp + (size_t)16 * D_ + dcn, D_, lane);
    v16h bn2 = load_bT_f16(Wp + (size_t)32 * D_ + dcn, D_, lane);
    v16h bn3 = load_bT_f16(Wp + (size_t)48 * D_ + dcn, D_, lane);
    // Consume current step (loads above are in flight).
    acc0 = wmma_f16(a, b0, acc0);
    acc1 = wmma_f16(a, b1, acc1);
    acc2 = wmma_f16(a, b2, acc2);
    acc3 = wmma_f16(a, b3, acc3);
    a = an; b0 = bn0; b1 = bn1; b2 = bn2; b3 = bn3;
  }
  // Epilogue
  acc0 = wmma_f16(a, b0, acc0);
  acc1 = wmma_f16(a, b1, acc1);
  acc2 = wmma_f16(a, b2, acc2);
  acc3 = wmma_f16(a, b3, acc3);

  // C/D layout: VGPR g -> row m = g + 8*(lane>=16), col n = lane&15
  int n  = lane & 15;
  int hf = lane >> 4;
  if (which == 2) {
    // V stored transposed: Vt[((b*H+h)*K + col) * S + s] -> contiguous-t frags later
    _Float16* vp = Vt + (size_t)(b * H_ + h) * K_ * S_;
    v8f* accs[4] = {&acc0, &acc1, &acc2, &acc3};
#pragma unroll
    for (int nt = 0; nt < 4; ++nt)
#pragma unroll
      for (int g = 0; g < 8; ++g) {
        int m = g + 8 * hf;
        vp[(size_t)(nt * 16 + n) * S_ + s0 + m] = (_Float16)((*accs[nt])[g]);
      }
  } else {
    _Float16* op = ((which == 0) ? Qf : Kf) + ((size_t)(b * H_ + h) * S_ + s0) * K_;
    v8f* accs[4] = {&acc0, &acc1, &acc2, &acc3};
#pragma unroll
    for (int nt = 0; nt < 4; ++nt)
#pragma unroll
      for (int g = 0; g < 8; ++g) {
        int m = g + 8 * hf;
        op[(size_t)m * K_ + nt * 16 + n] = (_Float16)((*accs[nt])[g]);
      }
  }
}

// ---------------- Kernel 2: causal flash attention --------------------------
// grid = (S/16, H, B), block = 32 (one wave owns a 16-query tile).
// K chunks double-buffered in LDS via async loads; V loads hidden under softmax.
__global__ __launch_bounds__(32)
void attn_kernel(const _Float16* __restrict__ Qf, const _Float16* __restrict__ Kf,
                 const _Float16* __restrict__ Vt, float* __restrict__ out) {
  __shared__ __align__(128) _Float16 kTile[2][32 * K_];  // 2 x 4KB K chunks
  __shared__ __align__(32)  _Float16 pTile[16 * 32];     // P relayout C->A

  int lane = threadIdx.x;
  int qt = blockIdx.x, h = blockIdx.y, b = blockIdx.z;
  int bh = b * H_ + h;

  const _Float16* Qp = Qf + ((size_t)bh * S_ + qt * 16) * K_;
  v16h qa0 = load_a_f16(Qp, K_, lane);       // d = 0..31
  v16h qa1 = load_a_f16(Qp + 32, K_, lane);  // d = 32..63

  float mrun[8], lrun[8];
#pragma unroll
  for (int g = 0; g < 8; ++g) { mrun[g] = -1e30f; lrun[g] = 0.0f; }
  v8f o0 = {}, o1 = {}, o2 = {}, o3 = {};

  int n  = lane & 15;
  int hf = lane >> 4;
  int k0 = hf * 16;
  const float scale = 0.125f;  // 1/sqrt(64)

  int qLast   = qt * 16 + 15;
  int nChunks = qLast / 32 + 1;   // causal: skip chunks entirely above diagonal

  const _Float16* Kbh = Kf + (size_t)bh * S_ * K_;
  const _Float16* Vbh = Vt + (size_t)bh * K_ * S_;
  unsigned la0 = (unsigned)(size_t)(const void*)&kTile[0][0] + (unsigned)lane * 16u;
  unsigned la1 = (unsigned)(size_t)(const void*)&kTile[1][0] + (unsigned)lane * 16u;

  // Prologue: stage chunk 0 into buffer 0.
  async_stage_4k(Kbh + lane * 8, la0);

  for (int c = 0; c < nChunks; ++c) {
    int tb = c * 32;
    int buf = c & 1;

    if (c + 1 < nChunks) {
      // Stage next chunk into the other buffer, then wait only for the
      // oldest 8 async ops (this chunk) -- next chunk streams in behind.
      async_stage_4k(Kbh + (size_t)(tb + 32) * K_ + lane * 8, buf ? la0 : la1);
      asm volatile("s_wait_asynccnt 0x8" ::: "memory");
    } else {
      asm volatile("s_wait_asynccnt 0x0" ::: "memory");
    }

    // S = Q * K^T from LDS-staged K chunk.
    const _Float16* kt0 = &kTile[buf][0] + n * K_ + k0;          // keys 0..15
    const _Float16* kt1 = &kTile[buf][0] + (16 + n) * K_ + k0;   // keys 16..31
    v8f sc0 = {}, sc1 = {};
    sc0 = wmma_f16(qa0, *(const v16h*)(kt0), sc0);
    sc0 = wmma_f16(qa1, *(const v16h*)(kt0 + 32), sc0);
    sc1 = wmma_f16(qa0, *(const v16h*)(kt1), sc1);
    sc1 = wmma_f16(qa1, *(const v16h*)(kt1 + 32), sc1);

    // Issue V fragment loads now; they are independent of the softmax below,
    // so their latency hides under ~100 VALU ops.
    const _Float16* Vp = Vbh + tb;                                // + d*S_ + t
    v16h vb0 = load_bT_f16(Vp,           S_, lane);
    v16h vb1 = load_bT_f16(Vp + 16 * S_, S_, lane);
    v16h vb2 = load_bT_f16(Vp + 32 * S_, S_, lane);
    v16h vb3 = load_bT_f16(Vp + 48 * S_, S_, lane);
    if (c + 1 < nChunks) {
      __builtin_prefetch(Vp + 32 + (size_t)lane * S_, 0, 0);      // next V chunk
      __builtin_prefetch(Vp + 32 + (size_t)(lane + 32) * S_, 0, 0);
    }

    // Online softmax per row (rows live in 16-lane groups -> xor 1,2,4,8)
#pragma unroll
    for (int g = 0; g < 8; ++g) {
      int m  = g + 8 * hf;
      int sq = qt * 16 + m;
      float v0 = (tb + n      <= sq) ? sc0[g] * scale : -1e30f;
      float v1 = (tb + 16 + n <= sq) ? sc1[g] * scale : -1e30f;
      float mx = fmaxf(v0, v1);
      mx = fmaxf(mx, __shfl_xor(mx, 1, 32));
      mx = fmaxf(mx, __shfl_xor(mx, 2, 32));
      mx = fmaxf(mx, __shfl_xor(mx, 4, 32));
      mx = fmaxf(mx, __shfl_xor(mx, 8, 32));
      float nm    = fmaxf(mrun[g], mx);
      float alpha = __expf(mrun[g] - nm);
      float p0 = __expf(v0 - nm);
      float p1 = __expf(v1 - nm);
      float rs = p0 + p1;
      rs += __shfl_xor(rs, 1, 32);
      rs += __shfl_xor(rs, 2, 32);
      rs += __shfl_xor(rs, 4, 32);
      rs += __shfl_xor(rs, 8, 32);
      lrun[g] = lrun[g] * alpha + rs;
      mrun[g] = nm;
      o0[g] *= alpha; o1[g] *= alpha; o2[g] *= alpha; o3[g] *= alpha;
      pTile[m * 32 + n]      = (_Float16)p0;
      pTile[m * 32 + 16 + n] = (_Float16)p1;
    }
    __syncthreads();

    // O += P(16x32) * V(32x64)
    v16h pa = load_a_f16((const _Float16*)pTile, 32, lane);
    o0 = wmma_f16(pa, vb0, o0);
    o1 = wmma_f16(pa, vb1, o1);
    o2 = wmma_f16(pa, vb2, o2);
    o3 = wmma_f16(pa, vb3, o3);
    __syncthreads();
  }

  // Normalize + store [B, S, H*K]
#pragma unroll
  for (int g = 0; g < 8; ++g) {
    int m  = g + 8 * hf;
    int sq = qt * 16 + m;
    float inv = 1.0f / lrun[g];
    float* op = out + ((size_t)b * S_ + sq) * (H_ * K_) + h * K_;
    op[ 0 + n] = o0[g] * inv;
    op[16 + n] = o1[g] * inv;
    op[32 + n] = o2[g] * inv;
    op[48 + n] = o3[g] * inv;
  }
}

extern "C" void kernel_launch(void* const* d_in, const int* in_sizes, int n_in,
                              void* d_out, int out_size, void* d_ws, size_t ws_size,
                              hipStream_t stream) {
  const float* x  = (const float*)d_in[0];
  const float* Wq = (const float*)d_in[1];
  const float* Wk = (const float*)d_in[2];
  const float* Wv = (const float*)d_in[3];
  float* out = (float*)d_out;

  size_t xhN   = (size_t)B_ * S_ * D_;           // 6.29M f16
  size_t wtN   = (size_t)3 * H_ * K_ * D_;       // 1.77M f16
  size_t perQK = (size_t)B_ * H_ * S_ * K_;      // 6.29M f16

  _Float16* xhp = (_Float16*)d_ws;               // total ws ~ 54.6 MB
  _Float16* Wtp = xhp + xhN;
  _Float16* Qf  = Wtp + wtN;
  _Float16* Kfp = Qf + perQK;
  _Float16* Vtp = Kfp + perQK;

  cvt_x_kernel<<<(unsigned)(xhN / 4 / 256), 256, 0, stream>>>(x, xhp);
  cvt_w_kernel<<<(unsigned)(wtN / 256), 256, 0, stream>>>(Wq, Wk, Wv, Wtp);

  dim3 g1((B_ * S_) / 64, H_, 3);
  proj_kernel<<<g1, 128, 0, stream>>>(xhp, Wtp, Qf, Kfp, Vtp);

  dim3 g2(S_ / 16, H_, B_);
  attn_kernel<<<g2, 32, 0, stream>>>(Qf, Kfp, Vtp, out);
}
